// Attention_9689446220613
// MI455X (gfx1250) — matrix operand order repaired and checked by reading, not compiled, not run on hardware
//
#include <hip/hip_runtime.h>

typedef __attribute__((ext_vector_type(16))) __bf16        v16bf;
typedef __attribute__((ext_vector_type(8)))  float         v8f;
typedef __attribute__((ext_vector_type(4)))  unsigned int  u32x4;

union FragBF {
    u32x4 u[2];
    v16bf v;
    unsigned short s[16];
};

union Pack8 {
    u32x4 u;
    unsigned short s[8];
};

__device__ __forceinline__ unsigned short f32_to_bf16(float f) {
    unsigned int u = __float_as_uint(f);
    unsigned int lsb = (u >> 16) & 1u;
    u += 0x7fffu + lsb;                 // round to nearest even
    return (unsigned short)(u >> 16);
}

// ---------------------------------------------------------------------------
// CDNA5 async global -> LDS copy (16B per lane, tracked by ASYNCcnt).
// Generic LDS pointer's low 32 bits are the LDS byte offset (LDS aperture).
// ---------------------------------------------------------------------------
__device__ __forceinline__ void async_copy_b128(void* lds_dst, const void* gsrc) {
    unsigned int lds_off = (unsigned int)(size_t)lds_dst;
    asm volatile("global_load_async_to_lds_b128 %0, %1, off"
                 :: "v"(lds_off), "v"(gsrc)
                 : "memory");
}

__device__ __forceinline__ void wait_async() {
#if __has_builtin(__builtin_amdgcn_s_wait_asynccnt)
    __builtin_amdgcn_s_wait_asynccnt(0);
#else
    asm volatile("s_wait_asynccnt 0x0" ::: "memory");
#endif
}

// ---------------------------------------------------------------------------
// f32 -> bf16 conversion (grid-stride)
// ---------------------------------------------------------------------------
__global__ void cvt_f32_bf16_kernel(const float* __restrict__ in,
                                    unsigned short* __restrict__ out, int n) {
    int i = blockIdx.x * blockDim.x + threadIdx.x;
    int stride = gridDim.x * blockDim.x;
    for (; i < n; i += stride) out[i] = f32_to_bf16(in[i]);
}

// ---------------------------------------------------------------------------
// bf16 GEMM:  out[M,N] = A[M,K] @ B[K,N] + bias[N]
// block tile 128x128, K-step 32, 256 threads = 8 waves (2x4 of 64x32 wave tiles)
// A tile staged via async global->LDS; B tile staged transposed via VGPRs.
// ---------------------------------------------------------------------------
template <bool OUT_BF16>
__global__ __launch_bounds__(256)
void gemm_bf16_kernel(const unsigned short* __restrict__ A,
                      const unsigned short* __restrict__ Bw,
                      const float* __restrict__ bias,
                      void* __restrict__ out,
                      int M, int N, int K)
{
    __shared__ alignas(16) unsigned short As[128 * 40];   // [m][k], pad 8
    __shared__ alignas(16) unsigned short Bs[128 * 40];   // [n][k] (transposed)

    const int tid   = threadIdx.x;
    const int lane  = tid & 31;
    const int wave  = tid >> 5;
    const int wm    = wave >> 2;        // 0..1
    const int wn    = wave & 3;         // 0..3
    const int lrow  = lane & 15;
    const int khalf = lane >> 4;

    const int bm = blockIdx.x * 128;
    const int bn = blockIdx.y * 128;

    v8f zero = {};
    v8f acc[4][2];
#pragma unroll
    for (int i = 0; i < 4; ++i)
#pragma unroll
        for (int j = 0; j < 2; ++j) acc[i][j] = zero;

    const int a_row = tid >> 1;          // 0..127
    const int a_col = (tid & 1) * 16;    // 0,16
    const int b_k   = tid >> 3;          // 0..31
    const int b_c   = (tid & 7) * 16;    // 0..112

    for (int k0 = 0; k0 < K; k0 += 32) {
        // ---- stage A tile (row-major) via async copies: 2 x 16B per thread ----
        {
            const unsigned short* ga = A + (size_t)(bm + a_row) * K + k0 + a_col;
            unsigned short* la = As + a_row * 40 + a_col;
            async_copy_b128(la,     ga);
            async_copy_b128(la + 8, ga + 8);
        }

        // ---- stage B tile transposed to [n][k] (must scatter -> VGPR path) ----
        const u32x4* gb = (const u32x4*)(Bw + (size_t)(k0 + b_k) * N + bn + b_c);
        Pack8 p0, p1;
        p0.u = gb[0];
        p1.u = gb[1];
#pragma unroll
        for (int j = 0; j < 8; ++j) Bs[(b_c + j) * 40 + b_k] = p0.s[j];
#pragma unroll
        for (int j = 0; j < 8; ++j) Bs[(b_c + 8 + j) * 40 + b_k] = p1.s[j];

        wait_async();
        __syncthreads();

        // ---- fragments ----
        FragBF afrag[4];
#pragma unroll
        for (int mt = 0; mt < 4; ++mt) {
            const unsigned short* pa = As + (wm * 64 + mt * 16 + lrow) * 40;
            afrag[mt].u[0] = *(const u32x4*)(pa + khalf * 8);
            afrag[mt].u[1] = *(const u32x4*)(pa + 16 + khalf * 8);
        }
        FragBF bfrag[2];
#pragma unroll
        for (int nt = 0; nt < 2; ++nt) {
            const unsigned short* pb = Bs + (wn * 32 + nt * 16 + lrow) * 40;
            bfrag[nt].u[0] = *(const u32x4*)(pb + khalf * 8);
            bfrag[nt].u[1] = *(const u32x4*)(pb + 16 + khalf * 8);
        }

#pragma unroll
        for (int mt = 0; mt < 4; ++mt)
#pragma unroll
            for (int nt = 0; nt < 2; ++nt)
                acc[mt][nt] = __builtin_amdgcn_wmma_f32_16x16x32_bf16(
                    false, afrag[mt].v, false, bfrag[nt].v,
                    (short)0, acc[mt][nt], false, false);
        __syncthreads();
    }

    // ---- epilogue ----
#pragma unroll
    for (int mt = 0; mt < 4; ++mt) {
#pragma unroll
        for (int nt = 0; nt < 2; ++nt) {
            const int col = bn + wn * 32 + nt * 16 + lrow;
            const float bv = bias[col];
#pragma unroll
            for (int r = 0; r < 8; ++r) {
                const int row = bm + wm * 64 + mt * 16 + r + khalf * 8;
                const float v = acc[mt][nt][r] + bv;
                if (OUT_BF16)
                    ((unsigned short*)out)[(size_t)row * N + col] = f32_to_bf16(v);
                else
                    ((float*)out)[(size_t)row * N + col] = v;
            }
        }
    }
}

// ---------------------------------------------------------------------------
// Flash attention over bf16 qkv buffer [B,T,3C].
// Block = (64 q rows) x (b,h); 128 threads = 4 waves x 16 q rows each.
// K tile staged via async global->LDS; V staged transposed via VGPRs.
// ---------------------------------------------------------------------------
__global__ __launch_bounds__(128)
void attention_kernel(const unsigned short* __restrict__ qkv,
                      unsigned short* __restrict__ attn_out)
{
    constexpr int T = 2048, C3 = 3072, Cc = 1024, D = 64;

    __shared__ alignas(16) unsigned short Ks[64 * 72];   // [key][d]
    __shared__ alignas(16) unsigned short Vs[64 * 72];   // [d][key] (transposed)
    __shared__ alignas(16) unsigned short Ps[64 * 72];   // [q][key]

    const int tid   = threadIdx.x;
    const int lane  = tid & 31;
    const int wave  = tid >> 5;
    const int lrow  = lane & 15;
    const int khalf = lane >> 4;

    const int qb = blockIdx.x;     // 0..31
    const int h  = blockIdx.y;     // 0..15
    const int b  = blockIdx.z;     // 0..1
    const int qbase = qb * 64;

    // Q fragments: A-layout, 16 rows x 64 d (two 16x32 chunks) straight from global
    FragBF qf[2];
    {
        const int t = qbase + wave * 16 + lrow;
        const unsigned short* qp = qkv + (size_t)(b * T + t) * C3 + h * D;
#pragma unroll
        for (int c = 0; c < 2; ++c) {
            qf[c].u[0] = *(const u32x4*)(qp + c * 32 + khalf * 8);
            qf[c].u[1] = *(const u32x4*)(qp + c * 32 + 16 + khalf * 8);
        }
    }

    v8f zero = {};
    v8f o[4];
#pragma unroll
    for (int dt = 0; dt < 4; ++dt) o[dt] = zero;
    float mrow[8], lsum[8];
#pragma unroll
    for (int r = 0; r < 8; ++r) { mrow[r] = -1e30f; lsum[r] = 0.0f; }

    const float kLog2e = 1.4426950408889634f;
    const float kScale = 0.03125f;          // 1/sqrt(C), C = 1024

    for (int j = 0; j <= qb; ++j) {
        const int kbase = j * 64;
        __syncthreads();   // previous iteration's LDS reads complete
        // ---- stage K (async, natural layout) and V (transposed via VGPRs) ----
        {
            const int key = tid >> 1;
            const int seg = (tid & 1) * 32;
            const unsigned short* kp =
                qkv + (size_t)(b * T + kbase + key) * C3 + Cc + h * D + seg;
            const unsigned short* vp = kp + Cc;
#pragma unroll
            for (int q4 = 0; q4 < 4; ++q4) {
                async_copy_b128(Ks + key * 72 + seg + q4 * 8, kp + q4 * 8);
                Pack8 pv;
                pv.u = *(const u32x4*)(vp + q4 * 8);
#pragma unroll
                for (int e = 0; e < 8; ++e)
                    Vs[(seg + q4 * 8 + e) * 72 + key] = pv.s[e];
            }
        }
        wait_async();
        __syncthreads();

        // ---- S = Q @ K^T : 16x64 per wave (4 tiles of 16x16) ----
        v8f s[4];
#pragma unroll
        for (int nt = 0; nt < 4; ++nt) {
            v8f acc = zero;
#pragma unroll
            for (int c = 0; c < 2; ++c) {
                FragBF bf;
                const unsigned short* pk = Ks + (nt * 16 + lrow) * 72 + c * 32;
                bf.u[0] = *(const u32x4*)(pk + khalf * 8);
                bf.u[1] = *(const u32x4*)(pk + 16 + khalf * 8);
                acc = __builtin_amdgcn_wmma_f32_16x16x32_bf16(
                    false, qf[c].v, false, bf.v, (short)0, acc, false, false);
            }
            s[nt] = acc;
        }

        // ---- scale + causal mask ----
#pragma unroll
        for (int nt = 0; nt < 4; ++nt) {
            const int ki = kbase + nt * 16 + lrow;
#pragma unroll
            for (int r = 0; r < 8; ++r) {
                const int qi = qbase + wave * 16 + r + khalf * 8;
                float v = s[nt][r] * kScale;
                if (ki > qi) v = -3.0e38f;
                s[nt][r] = v;
            }
        }

        // ---- online softmax (row stats via 16-lane shfl reductions) ----
        float alpha[8];
#pragma unroll
        for (int r = 0; r < 8; ++r) {
            float rm = fmaxf(fmaxf(s[0][r], s[1][r]), fmaxf(s[2][r], s[3][r]));
#pragma unroll
            for (int m = 1; m < 16; m <<= 1)
                rm = fmaxf(rm, __shfl_xor(rm, m, 32));
            const float mnew = fmaxf(mrow[r], rm);
            alpha[r] = exp2f((mrow[r] - mnew) * kLog2e);
            mrow[r] = mnew;
            float rs = 0.0f;
#pragma unroll
            for (int nt = 0; nt < 4; ++nt) {
                const float p = exp2f((s[nt][r] - mnew) * kLog2e);
                s[nt][r] = p;
                rs += p;
            }
#pragma unroll
            for (int m = 1; m < 16; m <<= 1)
                rs += __shfl_xor(rs, m, 32);
            lsum[r] = lsum[r] * alpha[r] + rs;
        }
#pragma unroll
        for (int dt = 0; dt < 4; ++dt)
#pragma unroll
            for (int r = 0; r < 8; ++r) o[dt][r] *= alpha[r];

        // ---- P (C-layout f32) -> LDS bf16 so it can be re-read as A-frags ----
#pragma unroll
        for (int nt = 0; nt < 4; ++nt)
#pragma unroll
            for (int r = 0; r < 8; ++r)
                Ps[(wave * 16 + r + khalf * 8) * 72 + nt * 16 + lrow] =
                    f32_to_bf16(s[nt][r]);
        __syncthreads();

        // ---- O += P @ V ----
#pragma unroll
        for (int c = 0; c < 2; ++c) {
            FragBF pf;
            const unsigned short* pp = Ps + (wave * 16 + lrow) * 72 + c * 32;
            pf.u[0] = *(const u32x4*)(pp + khalf * 8);
            pf.u[1] = *(const u32x4*)(pp + 16 + khalf * 8);
#pragma unroll
            for (int dt = 0; dt < 4; ++dt) {
                FragBF vf;
                const unsigned short* pv = Vs + (dt * 16 + lrow) * 72 + c * 32;
                vf.u[0] = *(const u32x4*)(pv + khalf * 8);
                vf.u[1] = *(const u32x4*)(pv + 16 + khalf * 8);
                o[dt] = __builtin_amdgcn_wmma_f32_16x16x32_bf16(
                    false, pf.v, false, vf.v, (short)0, o[dt], false, false);
            }
        }
    }

    // ---- normalize, write [B,T,C] bf16 ----
#pragma unroll
    for (int r = 0; r < 8; ++r) lsum[r] = 1.0f / lsum[r];
#pragma unroll
    for (int dt = 0; dt < 4; ++dt) {
#pragma unroll
        for (int r = 0; r < 8; ++r) {
            const int t   = qbase + wave * 16 + r + khalf * 8;
            const int col = h * D + dt * 16 + lrow;
            attn_out[(size_t)(b * T + t) * Cc + col] =
                f32_to_bf16(o[dt][r] * lsum[r]);
        }
    }
}

// ---------------------------------------------------------------------------
extern "C" void kernel_launch(void* const* d_in, const int* in_sizes, int n_in,
                              void* d_out, int out_size, void* d_ws, size_t ws_size,
                              hipStream_t stream) {
    (void)in_sizes; (void)n_in; (void)out_size; (void)ws_size;

    const float* x      = (const float*)d_in[0];
    const float* w_qkv  = (const float*)d_in[1];
    const float* b_qkv  = (const float*)d_in[2];
    const float* w_proj = (const float*)d_in[3];
    const float* b_proj = (const float*)d_in[4];

    const int Bb = 2, T = 2048, Cc = 1024;
    const size_t nx   = (size_t)Bb * T * Cc;        // 4,194,304
    const size_t nwq  = (size_t)Cc * 3 * Cc;        // 3,145,728
    const size_t nwp  = (size_t)Cc * Cc;            // 1,048,576
    const size_t nqkv = (size_t)Bb * T * 3 * Cc;    // 12,582,912

    unsigned short* xb   = (unsigned short*)d_ws;
    unsigned short* wqb  = xb + nx;
    unsigned short* wpb  = wqb + nwq;
    unsigned short* qkvb = wpb + nwp;
    unsigned short* attb = qkvb + nqkv;              // total ~48 MB of ws

    cvt_f32_bf16_kernel<<<2048, 256, 0, stream>>>(x, xb, (int)nx);
    cvt_f32_bf16_kernel<<<2048, 256, 0, stream>>>(w_qkv, wqb, (int)nwq);
    cvt_f32_bf16_kernel<<<1024, 256, 0, stream>>>(w_proj, wpb, (int)nwp);

    // qkv = x @ w_qkv + b_qkv   (M=4096, N=3072, K=1024), bf16 out
    gemm_bf16_kernel<true><<<dim3(32, 24), 256, 0, stream>>>(
        xb, wqb, b_qkv, qkvb, 4096, 3072, 1024);

    // flash attention: qkv -> attn (bf16 [B,T,C])
    attention_kernel<<<dim3(32, 16, 2), 128, 0, stream>>>(qkvb, attb);

    // out = attn @ w_proj + b_proj   (M=4096, N=1024, K=1024), f32 out
    gemm_bf16_kernel<false><<<dim3(32, 8), 256, 0, stream>>>(
        attb, wpb, b_proj, d_out, 4096, 1024, 1024);
}
